// SlotAttention_63848983822622
// MI455X (gfx1250) — matrix operand order repaired and checked by reading, not compiled, not run on hardware
//
#include <hip/hip_runtime.h>
#include <math.h>

// ---------------------------------------------------------------------------
// Types for CDNA5 WMMA (wave32): v_wmma_f32_16x16x32_bf16
// ---------------------------------------------------------------------------
typedef __bf16 bh16;
typedef __attribute__((ext_vector_type(16))) __bf16 v16bf;
typedef __attribute__((ext_vector_type(8)))  __bf16 v8bf;
typedef __attribute__((ext_vector_type(8)))  float  v8f;

union V16 { v16bf v; v8bf h[2]; };

__device__ __forceinline__ v8f wmma_bf16(v16bf a, v16bf b, v8f c) {
  // 8 args: (neg_a, A, neg_b, B, c_mod, C, reuse_a, reuse_b)
  return __builtin_amdgcn_wmma_f32_16x16x32_bf16(false, a, false, b, (short)0, c, false, false);
}

__device__ __forceinline__ bh16 f2bf(float f) {
  union { unsigned int u; float f; } c; c.f = f;
  unsigned int u = c.u + 0x7FFFu + ((c.u >> 16) & 1u);   // round-to-nearest-even
  union { unsigned short s; __bf16 b; } r; r.s = (unsigned short)(u >> 16);
  return r.b;
}

__device__ __forceinline__ float sigm(float x) { return 1.f / (1.f + __expf(-x)); }

// ---------------------------------------------------------------------------
// Tiny prep: transpose + cast a 256x256 f32 weight to [N,K] bf16
// ---------------------------------------------------------------------------
__global__ void k_transpose_cast(const float* __restrict__ W, bh16* __restrict__ WT) {
  int n = blockIdx.x;           // 256
  int k = threadIdx.x;          // 256
  WT[n * 256 + k] = f2bf(W[k * 256 + n]);
}

// ---------------------------------------------------------------------------
// Row LayerNorm over 256 cols; optional f32 and/or bf16 outputs.
// grid.x = rows, blockDim = 256
// ---------------------------------------------------------------------------
__global__ void k_ln_rows(const float* __restrict__ X, const float* __restrict__ g,
                          const float* __restrict__ b, float* __restrict__ outf,
                          bh16* __restrict__ outb) {
  __shared__ float red[256];
  const int  tid = threadIdx.x;
  const long row = blockIdx.x;
  float x = X[row * 256 + tid];
  red[tid] = x; __syncthreads();
  #pragma unroll
  for (int s = 128; s > 0; s >>= 1) { if (tid < s) red[tid] += red[tid + s]; __syncthreads(); }
  float mean = red[0] * (1.f / 256.f); __syncthreads();
  float d = x - mean;
  red[tid] = d * d; __syncthreads();
  #pragma unroll
  for (int s = 128; s > 0; s >>= 1) { if (tid < s) red[tid] += red[tid + s]; __syncthreads(); }
  float var = red[0] * (1.f / 256.f);
  float y = d * rsqrtf(var + 1e-5f) * g[tid] + b[tid];
  if (outf) outf[row * 256 + tid] = y;
  if (outb) outb[row * 256 + tid] = f2bf(y);
}

// ---------------------------------------------------------------------------
// Projection GEMM: D[Mr,256] = A[Mr,K] @ WT[N=256,K]^T + bias  (all bf16 in, bf16 out)
// OUT_T==0: out[m*ldOut + n], OUT_T==1: out[n*ldOut + m] (transposed store)
// grid.x = Mr/16, blockDim = 128 (4 waves, wave w covers n in [64w, 64w+64))
// ---------------------------------------------------------------------------
template <int OUT_T>
__global__ __launch_bounds__(128)
void k_gemm_proj(const bh16* __restrict__ A, const bh16* __restrict__ WT,
                 const float* __restrict__ bias, bh16* __restrict__ out,
                 int K, long ldOut) {
  const int lane = threadIdx.x & 31;
  const int wave = threadIdx.x >> 5;
  const int half = lane >> 4;
  const int lm   = lane & 15;
  const long m0  = (long)blockIdx.x * 16;
  const int nbase = wave * 64;
  v8f acc[4] = {};
  const bh16* arow = A + (m0 + lm) * (long)K;
  for (int kk = 0; kk < K; kk += 32) {
    V16 av;
    av.h[0] = *(const v8bf*)(arow + kk + 16 * half);
    av.h[1] = *(const v8bf*)(arow + kk + 16 * half + 8);
    #pragma unroll
    for (int t = 0; t < 4; ++t) {
      const bh16* brow = WT + (long)(nbase + t * 16 + lm) * K + kk + 16 * half;
      V16 bv;
      bv.h[0] = *(const v8bf*)(brow);
      bv.h[1] = *(const v8bf*)(brow + 8);
      acc[t] = wmma_bf16(av.v, bv.v, acc[t]);
    }
  }
  #pragma unroll
  for (int t = 0; t < 4; ++t) {
    int n = nbase + t * 16 + lm;
    float bn = bias ? bias[n] : 0.f;
    #pragma unroll
    for (int r = 0; r < 8; ++r) {
      long m = m0 + r + 8 * half;
      float v = acc[t][r] + bn;
      if (OUT_T == 0) out[m * ldOut + n] = f2bf(v);
      else            out[(long)n * ldOut + m] = f2bf(v);
    }
  }
}

// ---------------------------------------------------------------------------
// logits[64, M] = (Q[64,256] @ Kbf[M,256]^T) * scale   (f32 out)
// grid.x = M/64, blockDim = 128; wave w handles m-tile blockIdx.x*64 + 16w,
// computing all 4 s-tiles (B reused across s-tiles; Q is L0-resident).
// ---------------------------------------------------------------------------
__global__ __launch_bounds__(128)
void k_logits(const bh16* __restrict__ Q, const bh16* __restrict__ Kb,
              float* __restrict__ logits, int Mtot, float scale) {
  const int lane = threadIdx.x & 31;
  const int wave = threadIdx.x >> 5;
  const int half = lane >> 4;
  const int lm   = lane & 15;
  const long m0  = (long)blockIdx.x * 64 + wave * 16;
  v8f acc[4] = {};
  for (int kk = 0; kk < 256; kk += 32) {
    const bh16* brow = Kb + (m0 + lm) * 256 + kk + 16 * half;
    V16 bv;
    bv.h[0] = *(const v8bf*)(brow);
    bv.h[1] = *(const v8bf*)(brow + 8);
    #pragma unroll
    for (int st = 0; st < 4; ++st) {
      const bh16* arow = Q + (st * 16 + lm) * 256 + kk + 16 * half;
      V16 av;
      av.h[0] = *(const v8bf*)(arow);
      av.h[1] = *(const v8bf*)(arow + 8);
      acc[st] = wmma_bf16(av.v, bv.v, acc[st]);
    }
  }
  #pragma unroll
  for (int st = 0; st < 4; ++st) {
    #pragma unroll
    for (int r = 0; r < 8; ++r) {
      int s = st * 16 + r + 8 * half;
      logits[(long)s * Mtot + m0 + lm] = acc[st][r] * scale;
    }
  }
}

// ---------------------------------------------------------------------------
// softmax over slots (axis 0) per column m; in-place on attn (f32), and
// writes a bf16 mirror for the WMMA A operand of the update GEMM.
// ---------------------------------------------------------------------------
__global__ void k_softmax(float* __restrict__ attn, bh16* __restrict__ attn_bf, int Mtot) {
  long m = (long)blockIdx.x * blockDim.x + threadIdx.x;
  float v[64];
  float mx = -3.4e38f;
  #pragma unroll
  for (int s = 0; s < 64; ++s) { float x = attn[(long)s * Mtot + m]; v[s] = x; mx = fmaxf(mx, x); }
  float sum = 0.f;
  #pragma unroll
  for (int s = 0; s < 64; ++s) { float e = __expf(v[s] - mx); v[s] = e; sum += e; }
  float inv = 1.f / sum;
  #pragma unroll
  for (int s = 0; s < 64; ++s) {
    float a = v[s] * inv;
    attn[(long)s * Mtot + m] = a;
    attn_bf[(long)s * Mtot + m] = f2bf(a);
  }
}

// ---------------------------------------------------------------------------
// up[64,256] = attn[64,M] @ V[M,256], V staged transposed VT[256,M] bf16.
// K split over grid.y; wave = s-tile; each wave covers 4 n-tiles so one A load
// feeds 4 WMMAs. Inner loop is explicitly double-buffered (prefetch k+32 while
// WMMAs consume k) so VMEM overlaps matrix ops within a wave. Results go to a
// per-slice partials buffer (coalesced stores, no atomics); k_reduce_up sums.
// grid = (4 n-groups, KSPLIT, 2), blockDim = 128.
// ---------------------------------------------------------------------------
__global__ __launch_bounds__(128)
void k_update_gemm(const bh16* __restrict__ attn,
                   const bh16* __restrict__ VT_in, const bh16* __restrict__ VT_tgt,
                   float* __restrict__ partials, int Mtot, int kchunk) {
  const bh16* BT = blockIdx.z ? VT_tgt : VT_in;
  const int lane = threadIdx.x & 31;
  const int wave = threadIdx.x >> 5;
  const int half = lane >> 4;
  const int lm   = lane & 15;
  const int s0 = wave * 16;
  const int n0 = blockIdx.x * 64;
  const long kb = (long)blockIdx.y * kchunk;

  const bh16* arow = attn + (long)(s0 + lm) * Mtot + kb;
  const bh16* bbase = BT + (long)(n0 + lm) * Mtot + kb + 8 * half;
  const long bstep = (long)16 * Mtot;   // stride between n-tiles

  v8f acc[4] = {};
  V16 av, bv[4];
  // prologue: load k-step 0
  av.h[0] = *(const v8bf*)(arow + 8 * half);
  av.h[1] = *(const v8bf*)(arow + 16 + 8 * half);
  #pragma unroll
  for (int t = 0; t < 4; ++t) {
    const bh16* p = bbase + t * bstep;
    bv[t].h[0] = *(const v8bf*)(p);
    bv[t].h[1] = *(const v8bf*)(p + 16);
  }
  for (int kk = 0; kk < kchunk - 32; kk += 32) {
    V16 an, bn[4];
    const int kn = kk + 32;
    an.h[0] = *(const v8bf*)(arow + kn + 8 * half);
    an.h[1] = *(const v8bf*)(arow + kn + 16 + 8 * half);
    #pragma unroll
    for (int t = 0; t < 4; ++t) {
      const bh16* p = bbase + t * bstep + kn;
      bn[t].h[0] = *(const v8bf*)(p);
      bn[t].h[1] = *(const v8bf*)(p + 16);
    }
    #pragma unroll
    for (int t = 0; t < 4; ++t) acc[t] = wmma_bf16(av.v, bv[t].v, acc[t]);
    av = an;
    #pragma unroll
    for (int t = 0; t < 4; ++t) bv[t] = bn[t];
  }
  #pragma unroll
  for (int t = 0; t < 4; ++t) acc[t] = wmma_bf16(av.v, bv[t].v, acc[t]);

  // epilogue: plain coalesced stores into this slice's partial [64,256] block
  float* pt = partials + (long)(blockIdx.z * gridDim.y + blockIdx.y) * (64 * 256);
  #pragma unroll
  for (int t = 0; t < 4; ++t) {
    int n = n0 + t * 16 + lm;
    #pragma unroll
    for (int r = 0; r < 8; ++r) {
      int s = s0 + r + 8 * half;
      pt[s * 256 + n] = acc[t][r];
    }
  }
}

// Sum KSPLIT partial [64,256] slices per matrix (z=0: in, z=1: tgt).
__global__ void k_reduce_up(const float* __restrict__ partials,
                            float* __restrict__ up_in, float* __restrict__ up_tgt,
                            int nslices) {
  int i = blockIdx.x * blockDim.x + threadIdx.x;   // 0 .. 2*16384
  int z = i >> 14;
  int e = i & 16383;
  const float* base = partials + (long)z * nslices * 16384;
  float s = 0.f;
  for (int k = 0; k < nslices; ++k) s += base[(long)k * 16384 + e];
  (z ? up_tgt : up_in)[e] = s;
}

// ---------------------------------------------------------------------------
// GRU: gi = x@Wih + bih, gh = h@Whh + bhh  (Wih/Whh: [256, 768] row-major)
// grid = (3, 64), blockDim = 256; x/h rows cached in LDS.
// ---------------------------------------------------------------------------
__global__ void k_gru_mm(const float* __restrict__ x, const float* __restrict__ h,
                         const float* __restrict__ Wih, const float* __restrict__ Whh,
                         const float* __restrict__ bih, const float* __restrict__ bhh,
                         float* __restrict__ gi, float* __restrict__ gh) {
  __shared__ float sx[256], sh[256];
  const int s = blockIdx.y;
  const int col = blockIdx.x * 256 + threadIdx.x;
  sx[threadIdx.x] = x[s * 256 + threadIdx.x];
  sh[threadIdx.x] = h[s * 256 + threadIdx.x];
  __syncthreads();
  float ai = bih[col], ah = bhh[col];
  for (int k = 0; k < 256; ++k) {
    ai += sx[k] * Wih[k * 768 + col];
    ah += sh[k] * Whh[k * 768 + col];
  }
  gi[s * 768 + col] = ai;
  gh[s * 768 + col] = ah;
}

__global__ void k_gru_combine(const float* __restrict__ gi, const float* __restrict__ gh,
                              float* __restrict__ h) {
  const int s = blockIdx.x, j = threadIdx.x;
  float r = sigm(gi[s * 768 + j]       + gh[s * 768 + j]);
  float z = sigm(gi[s * 768 + 256 + j] + gh[s * 768 + 256 + j]);
  float n = tanhf(gi[s * 768 + 512 + j] + r * gh[s * 768 + 512 + j]);
  float ho = h[s * 256 + j];
  h[s * 256 + j] = (1.f - z) * n + z * ho;
}

// ---------------------------------------------------------------------------
// Small scalar GEMM for the slot MLPs: out[s,n] = (resid?) + act(A[s,:]@W[:,n] + b[n])
// grid = (ceil(N/blockDim), 64); A row cached in dynamic LDS (K floats).
// ---------------------------------------------------------------------------
__global__ void k_mm_scalar(const float* __restrict__ A, const float* __restrict__ W,
                            const float* __restrict__ bias, const float* __restrict__ resid,
                            float* __restrict__ out, int K, int N, int relu) {
  extern __shared__ float sa[];
  const int s = blockIdx.y;
  for (int k = threadIdx.x; k < K; k += blockDim.x) sa[k] = A[s * K + k];
  __syncthreads();
  const int n = blockIdx.x * blockDim.x + threadIdx.x;
  if (n >= N) return;
  float acc = bias[n];
  for (int k = 0; k < K; ++k) acc += sa[k] * W[k * N + n];
  if (relu) acc = fmaxf(acc, 0.f);
  if (resid) acc += resid[s * N + n];
  out[s * N + n] = acc;
}

__global__ void k_copy(const float* __restrict__ src, float* __restrict__ dst, int n) {
  int i = blockIdx.x * blockDim.x + threadIdx.x;
  if (i < n) dst[i] = src[i];
}

// ---------------------------------------------------------------------------
// Host-side orchestration
// ---------------------------------------------------------------------------
extern "C" void kernel_launch(void* const* d_in, const int* in_sizes, int n_in,
                              void* d_out, int out_size, void* d_ws, size_t ws_size,
                              hipStream_t stream) {
  constexpr int M = 65536, S = 64, D = 256;
  constexpr int KSPLIT = 32;

  const float* inputs     = (const float*)d_in[0];
  const float* target     = (const float*)d_in[1];
  const float* in_slots0  = (const float*)d_in[2];
  const float* tgt_slots0 = (const float*)d_in[3];
  const float* ln_in_g  = (const float*)d_in[4];  const float* ln_in_b  = (const float*)d_in[5];
  const float* ln_tgt_g = (const float*)d_in[6];  const float* ln_tgt_b = (const float*)d_in[7];
  const float* Wk  = (const float*)d_in[8];  const float* bk  = (const float*)d_in[9];
  const float* Wv  = (const float*)d_in[10]; const float* bv  = (const float*)d_in[11];
  const float* Wvt = (const float*)d_in[12]; const float* bvt = (const float*)d_in[13];
  const float* ln_s_g = (const float*)d_in[14]; const float* ln_s_b = (const float*)d_in[15];
  const float* Wq  = (const float*)d_in[16]; const float* bq  = (const float*)d_in[17];
  const float* giWih = (const float*)d_in[18]; const float* giWhh = (const float*)d_in[19];
  const float* gibih = (const float*)d_in[20]; const float* gibhh = (const float*)d_in[21];
  const float* gtWih = (const float*)d_in[22]; const float* gtWhh = (const float*)d_in[23];
  const float* gtbih = (const float*)d_in[24]; const float* gtbhh = (const float*)d_in[25];
  const float* ln_mi_g = (const float*)d_in[26]; const float* ln_mi_b = (const float*)d_in[27];
  const float* miW1 = (const float*)d_in[28]; const float* mib1 = (const float*)d_in[29];
  const float* miW2 = (const float*)d_in[30]; const float* mib2 = (const float*)d_in[31];
  const float* ln_mt_g = (const float*)d_in[32]; const float* ln_mt_b = (const float*)d_in[33];
  const float* mtW1 = (const float*)d_in[34]; const float* mtb1 = (const float*)d_in[35];
  const float* mtW2 = (const float*)d_in[36]; const float* mtb2 = (const float*)d_in[37];

  // workspace carve-out (~182 MB)
  char* ws = (char*)d_ws;
  size_t off = 0;
  auto take = [&](size_t bytes) -> char* {
    char* p = ws + off; off += (bytes + 255) & ~(size_t)255; return p;
  };
  bh16* XNbf   = (bh16*)take((size_t)M * D * 2);   // LN(inputs) bf16
  bh16* TNbf   = (bh16*)take((size_t)M * D * 2);   // LN(target) bf16
  bh16* Kbf    = (bh16*)take((size_t)M * D * 2);   // in_k, row-major [M,256]
  bh16* VbfT   = (bh16*)take((size_t)M * D * 2);   // in_v^T [256,M]
  bh16* VtbfT  = (bh16*)take((size_t)M * D * 2);   // tgt_v^T [256,M]
  bh16* attnbf = (bh16*)take((size_t)S * M * 2);   // attn bf16 mirror [64,M]
  bh16* WkT  = (bh16*)take((size_t)D * D * 2);
  bh16* WvT  = (bh16*)take((size_t)D * D * 2);
  bh16* WvtT = (bh16*)take((size_t)D * D * 2);
  bh16* WqT  = (bh16*)take((size_t)D * D * 2);
  bh16* qbf     = (bh16*)take((size_t)S * D * 2);
  bh16* slotsNb = (bh16*)take((size_t)S * D * 2);
  float* lnbuf  = (float*)take((size_t)S * D * 4);
  float* in_cur  = (float*)take((size_t)S * D * 4);
  float* tgt_cur = (float*)take((size_t)S * D * 4);
  float* up_in   = (float*)take((size_t)2 * S * D * 4);  // up_in | up_tgt adjacent
  float* up_tgt  = up_in + S * D;
  float* partials = (float*)take((size_t)2 * KSPLIT * S * D * 4); // 4 MB
  float* gi = (float*)take((size_t)S * 768 * 4);
  float* gh = (float*)take((size_t)S * 768 * 4);
  float* h1 = (float*)take((size_t)S * D * 4);
  (void)ws_size; (void)in_sizes; (void)n_in; (void)out_size;

  float* out_in   = (float*)d_out;
  float* out_tgt  = out_in + S * D;
  float* attn_out = out_in + 2 * S * D;   // [64, M] f32, written each iter; last iter persists

  // ---- prep: weight transpose+cast, input LayerNorms, K/V/Vt projections ----
  k_transpose_cast<<<256, 256, 0, stream>>>(Wk,  WkT);
  k_transpose_cast<<<256, 256, 0, stream>>>(Wv,  WvT);
  k_transpose_cast<<<256, 256, 0, stream>>>(Wvt, WvtT);
  k_transpose_cast<<<256, 256, 0, stream>>>(Wq,  WqT);

  k_ln_rows<<<M, 256, 0, stream>>>(inputs, ln_in_g,  ln_in_b,  nullptr, XNbf);
  k_ln_rows<<<M, 256, 0, stream>>>(target, ln_tgt_g, ln_tgt_b, nullptr, TNbf);

  k_gemm_proj<0><<<M / 16, 128, 0, stream>>>(XNbf, WkT,  bk,  Kbf,   D, (long)D);
  k_gemm_proj<1><<<M / 16, 128, 0, stream>>>(XNbf, WvT,  bv,  VbfT,  D, (long)M);
  k_gemm_proj<1><<<M / 16, 128, 0, stream>>>(TNbf, WvtT, bvt, VtbfT, D, (long)M);

  k_copy<<<(S * D) / 256, 256, 0, stream>>>(in_slots0,  in_cur,  S * D);
  k_copy<<<(S * D) / 256, 256, 0, stream>>>(tgt_slots0, tgt_cur, S * D);

  // ---- 3 slot-attention iterations ----
  for (int it = 0; it < 3; ++it) {
    // q = LN(in_slots) @ Wq + bq
    k_ln_rows<<<S, 256, 0, stream>>>(in_cur, ln_s_g, ln_s_b, nullptr, slotsNb);
    k_gemm_proj<0><<<S / 16, 128, 0, stream>>>(slotsNb, WqT, bq, qbf, D, (long)D);

    // logits + softmax over slots (written into the attn output region)
    k_logits<<<M / 64, 128, 0, stream>>>(qbf, Kbf, attn_out, M, 1.f / 16.f);
    k_softmax<<<M / 256, 256, 0, stream>>>(attn_out, attnbf, M);

    // in_up / tgt_up = attn @ {in_v, tgt_v}; K=65536 split 32-way, then reduce
    k_update_gemm<<<dim3(4, KSPLIT, 2), 128, 0, stream>>>(
        attnbf, VbfT, VtbfT, partials, M, M / KSPLIT);
    k_reduce_up<<<(2 * S * D) / 256, 256, 0, stream>>>(partials, up_in, up_tgt, KSPLIT);

    // GRU cells (in-place slot update)
    k_gru_mm<<<dim3(3, S), 256, 0, stream>>>(up_in, in_cur, giWih, giWhh, gibih, gibhh, gi, gh);
    k_gru_combine<<<S, 256, 0, stream>>>(gi, gh, in_cur);
    k_gru_mm<<<dim3(3, S), 256, 0, stream>>>(up_tgt, tgt_cur, gtWih, gtWhh, gtbih, gtbhh, gi, gh);
    k_gru_combine<<<S, 256, 0, stream>>>(gi, gh, tgt_cur);

    // residual MLP (in): 256 -> 128 -> 256
    k_ln_rows<<<S, 256, 0, stream>>>(in_cur, ln_mi_g, ln_mi_b, lnbuf, nullptr);
    k_mm_scalar<<<dim3(1, S), 128, 256 * 4, stream>>>(lnbuf, miW1, mib1, nullptr, h1, 256, 128, 1);
    k_mm_scalar<<<dim3(2, S), 128, 128 * 4, stream>>>(h1, miW2, mib2, in_cur, in_cur, 128, 256, 0);

    // residual MLP (tgt): 256 -> 256 -> 256
    k_ln_rows<<<S, 256, 0, stream>>>(tgt_cur, ln_mt_g, ln_mt_b, lnbuf, nullptr);
    k_mm_scalar<<<dim3(2, S), 128, 256 * 4, stream>>>(lnbuf, mtW1, mtb1, nullptr, h1, 256, 256, 1);
    k_mm_scalar<<<dim3(2, S), 128, 256 * 4, stream>>>(h1, mtW2, mtb2, tgt_cur, tgt_cur, 256, 256, 0);
  }

  // ---- emit slot outputs (attn already in place from last iteration) ----
  k_copy<<<(S * D) / 256, 256, 0, stream>>>(in_cur,  out_in,  S * D);
  k_copy<<<(S * D) / 256, 256, 0, stream>>>(tgt_cur, out_tgt, S * D);
}
